// MetaNETS_31748398252424
// MI455X (gfx1250) — matrix operand order repaired and checked by reading, not compiled, not run on hardware
//
#include <hip/hip_runtime.h>
#include <hip/hip_bf16.h>
#include <math.h>

typedef _Float16 f16;
typedef __attribute__((ext_vector_type(16))) _Float16 v16h;
typedef __attribute__((ext_vector_type(8)))  float        v8f;
typedef __attribute__((ext_vector_type(4)))  unsigned int v4u;

#define BTOT   2048
#define NCTX   64
#define HD     128
#define ZD     64
#define RD     128
#define NSTEPS 20
#define LDW    136   // padded f16 row stride for transposed weights (16B aligned, bank-skewed)

union FragU { v4u q[2]; v16h h; };

struct Smem {
  // ---- f16 blocks (all offsets multiples of 16B) ----
  f16 WTd1[HD * 68];      // Wd1^T  [h=128][k=68] (66 used: 64 z + 2 x)
  f16 WTa [HD * LDW];     // fwd B operand: Wd2^T [out][in]   (encoder: We2^T)
  f16 WTb [HD * LDW];     // bwd B operand: Wd2 row-major pad (encoder: We3^T)
  f16 Wf1 [193 * HD];
  f16 Wf2 [HD * HD];
  f16 Wf3 [HD * ZD];
  f16 Aact[NCTX * HD];    // d1_act / e1_act / dd2  (A operand)
  f16 Apre[NCTX * HD];    // silu'(d1_pre)
  f16 Bpre[NCTX * HD];    // d2_pre / e2_act
  // ---- f32 small state ----
  float We1[3 * HD];
  float Wd3[HD];
  float be1[HD], be2[HD], be3[HD];
  float bd1[HD], bd2[HD];
  float bf1[HD], bf2[HD], bf3[ZD];
  float xs[NCTX * 2], ys[NCTX], ms[NCTX];
  float z[ZD], zrow[HD], S[HD], r[RD];
  float f1a[HD], f2a[HD], decp[NCTX * 4], gv[ZD];
  float bd3, msum;
};
static_assert(sizeof(Smem) <= 320 * 1024, "LDS overflow");

// ---------- helpers ----------
__device__ __forceinline__ float sigm(float x) { return 1.f / (1.f + __expf(-x)); }
__device__ __forceinline__ float siluf(float x) { return x * sigm(x); }
__device__ __forceinline__ float dsilu(float x) { float s = sigm(x); return s * (1.f + x * (1.f - s)); }

__device__ __forceinline__ void cp32(float* dst, const float* src, int n) {
  for (int i = threadIdx.x; i < n; i += 256) dst[i] = src[i];
}
__device__ __forceinline__ void cp16(f16* dst, const float* src, int n) {
  for (int i = threadIdx.x; i < n; i += 256) dst[i] = (f16)src[i];
}
// dst[o*ld + i] = src[i*O + o]  (transpose; global reads coalesced over o)
__device__ __forceinline__ void cpT(f16* dst, const float* src, int I, int O, int ld) {
  for (int idx = threadIdx.x; idx < I * O; idx += 256) {
    int i = idx / O, o = idx - i * O;
    dst[o * ld + i] = (f16)src[idx];
  }
}
// row-major copy with padded stride
__device__ __forceinline__ void cpPad(f16* dst, const float* src, int R, int C, int ld) {
  for (int idx = threadIdx.x; idx < R * C; idx += 256) {
    int rr = idx / C, cc = idx - rr * C;
    dst[rr * ld + cc] = (f16)src[idx];
  }
}

// Wave-cooperative [64,128] x [128,128] matmul: each wave owns m-tile (wave&3)
// and 4 n-tiles ((wave>>2)*4 ..+3).  A: row-major [64][128] f16.
// Bsrc[n][k] layout with row stride ldB supplies B element (k, n).
__device__ __forceinline__ void wave_mm(const f16* A, const f16* Bsrc, int ldB, v8f acc[4]) {
  const int lane = threadIdx.x & 31;
  const int wv   = threadIdx.x >> 5;
  const int mt   = wv & 3;
  const int ntB  = (wv >> 2) * 4;
  const int lm   = lane & 15;
  const int hs   = lane >> 4;
#pragma unroll
  for (int kc = 0; kc < 4; ++kc) {
    FragU a;
    const f16* ar = A + (mt * 16 + lm) * HD + kc * 32;
    a.q[0] = *(const v4u*)(ar + hs * 8);        // VGPR0-3: k = 8*hs + 0..7
    a.q[1] = *(const v4u*)(ar + 16 + hs * 8);   // VGPR4-7: k = 16 + 8*hs + 0..7
#pragma unroll
    for (int j = 0; j < 4; ++j) {
      FragU bf;
      const f16* br = Bsrc + ((ntB + j) * 16 + lm) * ldB + kc * 32;
      bf.q[0] = *(const v4u*)(br + hs * 8);       // k = 8*hs + 0..7
      bf.q[1] = *(const v4u*)(br + 16 + hs * 8);  // k = 16 + 8*hs + 0..7
      acc[j] = __builtin_amdgcn_wmma_f32_16x16x32_f16(false, a.h, false, bf.h,
                                                      (short)0, acc[j], false, false);
    }
  }
}

__global__ __launch_bounds__(256, 1) void metanets_persistent_kernel(
    const float* __restrict__ x_ctx, const float* __restrict__ y_ctx,
    const float* __restrict__ mask,  const float* __restrict__ z0,
    const float* __restrict__ noises,
    const float* __restrict__ gWe1, const float* __restrict__ gbe1,
    const float* __restrict__ gWe2, const float* __restrict__ gbe2,
    const float* __restrict__ gWe3, const float* __restrict__ gbe3,
    const float* __restrict__ gWd1, const float* __restrict__ gbd1,
    const float* __restrict__ gWd2, const float* __restrict__ gbd2,
    const float* __restrict__ gWd3, const float* __restrict__ gbd3,
    const float* __restrict__ gWf1, const float* __restrict__ gbf1,
    const float* __restrict__ gWf2, const float* __restrict__ gbf2,
    const float* __restrict__ gWf3, const float* __restrict__ gbf3,
    float* __restrict__ out)
{
  __shared__ Smem sm;
  const int b    = blockIdx.x;
  const int tid  = threadIdx.x;
  const int lane = tid & 31;
  const int wv   = tid >> 5;
  const int mt   = wv & 3;
  const int ntB  = (wv >> 2) * 4;
  const int lm   = lane & 15;
  const int hs   = lane >> 4;
  const v8f vzero = {0.f, 0.f, 0.f, 0.f, 0.f, 0.f, 0.f, 0.f};

  // ---------------- stage 0: load weights / per-batch data ----------------
  cpT(sm.WTd1, gWd1, 66, HD, 68);
  cpT(sm.WTa, gWe2, HD, HD, LDW);   // encoder weights first (slot reuse)
  cpT(sm.WTb, gWe3, HD, HD, LDW);
  cp16(sm.Wf1, gWf1, 193 * HD);
  cp16(sm.Wf2, gWf2, HD * HD);
  cp16(sm.Wf3, gWf3, HD * ZD);
  cp32(sm.We1, gWe1, 3 * HD);
  cp32(sm.Wd3, gWd3, HD);
  cp32(sm.be1, gbe1, HD); cp32(sm.be2, gbe2, HD); cp32(sm.be3, gbe3, HD);
  cp32(sm.bd1, gbd1, HD); cp32(sm.bd2, gbd2, HD);
  cp32(sm.bf1, gbf1, HD); cp32(sm.bf2, gbf2, HD); cp32(sm.bf3, gbf3, ZD);
  cp32(sm.xs, x_ctx + (size_t)b * NCTX * 2, NCTX * 2);
  cp32(sm.ys, y_ctx + (size_t)b * NCTX, NCTX);
  cp32(sm.ms, mask  + (size_t)b * NCTX, NCTX);
  cp32(sm.z,  z0    + (size_t)b * ZD,  ZD);
  if (tid < HD) sm.S[tid] = 0.f;
  if (tid == 0) { sm.bd3 = gbd3[0]; sm.msum = 0.f; }
  __syncthreads();

  // ---------------- encoder: r = maskedMean(MLP(x,y)) ----------------
  for (int idx = tid; idx < NCTX * HD; idx += 256) {
    int n = idx >> 7, h = idx & 127;
    float pre = sm.be1[h] + sm.xs[2 * n] * sm.We1[h]
              + sm.xs[2 * n + 1] * sm.We1[HD + h] + sm.ys[n] * sm.We1[2 * HD + h];
    sm.Aact[idx] = (f16)siluf(pre);
  }
  if (tid < NCTX) atomicAdd(&sm.msum, sm.ms[tid]);
  __syncthreads();
  { // e2 = silu(e1 @ We2 + be2)
    v8f acc[4] = {vzero, vzero, vzero, vzero};
    wave_mm(sm.Aact, sm.WTa, LDW, acc);
#pragma unroll
    for (int j = 0; j < 4; ++j) {
      int n = (ntB + j) * 16 + lm;
#pragma unroll
      for (int v = 0; v < 8; ++v) {
        int m = mt * 16 + v + 8 * hs;
        sm.Bpre[m * HD + n] = (f16)siluf(acc[j][v] + sm.be2[n]);
      }
    }
  }
  __syncthreads();
  { // e3 = e2 @ We3 + be3 ; rsum[n] += e3[m][n]*mask[m]
    v8f acc[4] = {vzero, vzero, vzero, vzero};
    wave_mm(sm.Bpre, sm.WTb, LDW, acc);
#pragma unroll
    for (int j = 0; j < 4; ++j) {
      int n = (ntB + j) * 16 + lm;
      float cs = 0.f;
#pragma unroll
      for (int v = 0; v < 8; ++v) {
        int m = mt * 16 + v + 8 * hs;
        cs += (acc[j][v] + sm.be3[n]) * sm.ms[m];
      }
      atomicAdd(&sm.S[n], cs);
    }
  }
  __syncthreads();
  if (tid < RD) { sm.r[tid] = sm.S[tid] / fmaxf(sm.msum, 1e-6f); sm.S[tid] = 0.f; }
  // overwrite encoder slots with decoder Wd2 (both orientations)
  cpT(sm.WTa, gWd2, HD, HD, LDW);       // B fwd:  element (k,n) = Wd2[k][n] = WTa[n][k]
  cpPad(sm.WTb, gWd2, HD, HD, LDW);     // B bwd:  element (k,n) of Wd2^T = Wd2[n][k] = WTb[n][k]
  __syncthreads();

  // ---------------- Langevin step loop ----------------
  const float dt  = 1.f / (float)NSTEPS;
  const float dif = sqrtf(2.f * dt);
  for (int st = 0; st < NSTEPS; ++st) {
    const float t = (float)st * dt;

    // P1: shared z-row of decoder layer1 (+ bias); S already zeroed
    if (tid < HD) {
      float a = sm.bd1[tid];
      const f16* wt = sm.WTd1 + tid * 68;
#pragma unroll 8
      for (int k = 0; k < ZD; ++k) a += sm.z[k] * (float)wt[k];
      sm.zrow[tid] = a;
    }
    __syncthreads();

    // P2: d1 act + silu' (layer1 forward, x-part per row)
    for (int idx = tid; idx < NCTX * HD; idx += 256) {
      int n = idx >> 7, h = idx & 127;
      const f16* wt = sm.WTd1 + h * 68;
      float pre = sm.zrow[h] + sm.xs[2 * n] * (float)wt[64] + sm.xs[2 * n + 1] * (float)wt[65];
      sm.Aact[idx] = (f16)siluf(pre);
      sm.Apre[idx] = (f16)dsilu(pre);
    }
    __syncthreads();

    // P3: d2_pre = d1_act @ Wd2 + bd2   (WMMA)
    {
      v8f acc[4] = {vzero, vzero, vzero, vzero};
      wave_mm(sm.Aact, sm.WTa, LDW, acc);
#pragma unroll
      for (int j = 0; j < 4; ++j) {
        int n = (ntB + j) * 16 + lm;
#pragma unroll
        for (int v = 0; v < 8; ++v) {
          int m = mt * 16 + v + 8 * hs;
          sm.Bpre[m * HD + n] = (f16)(acc[j][v] + sm.bd2[n]);
        }
      }
    }
    __syncthreads();

    // P4: dec[n] = silu(d2_pre) . Wd3 + bd3 ;  e[n] = (dec - y)*mask
    {
      int n = tid >> 2, p = tid & 3;
      float a = 0.f;
      for (int h = p * 32; h < p * 32 + 32; ++h)
        a += siluf((float)sm.Bpre[n * HD + h]) * sm.Wd3[h];
      sm.decp[n * 4 + p] = a;
    }
    __syncthreads();
    if (tid < NCTX) {
      float dec = sm.decp[tid * 4] + sm.decp[tid * 4 + 1] + sm.decp[tid * 4 + 2]
                + sm.decp[tid * 4 + 3] + sm.bd3;
      sm.decp[tid * 4] = (dec - sm.ys[tid]) * sm.ms[tid];  // e[n]
    }
    __syncthreads();

    // P5: dd2 = e[n] * Wd3[h] * silu'(d2_pre)   (A operand for backward)
    for (int idx = tid; idx < NCTX * HD; idx += 256) {
      int n = idx >> 7, h = idx & 127;
      sm.Aact[idx] = (f16)(sm.decp[n * 4] * sm.Wd3[h] * dsilu((float)sm.Bpre[idx]));
    }
    __syncthreads();

    // P6: dd1 = (dd2 @ Wd2^T) * silu'(d1_pre); column-reduce into S  (WMMA)
    {
      v8f acc[4] = {vzero, vzero, vzero, vzero};
      wave_mm(sm.Aact, sm.WTb, LDW, acc);
#pragma unroll
      for (int j = 0; j < 4; ++j) {
        int n = (ntB + j) * 16 + lm;
        float cs = 0.f;
#pragma unroll
        for (int v = 0; v < 8; ++v) {
          int m = mt * 16 + v + 8 * hs;
          cs += acc[j][v] * (float)sm.Apre[m * HD + n];
        }
        atomicAdd(&sm.S[n], cs);
      }
    }
    __syncthreads();

    // P7a: drift f1 (threads 0..127) || grad g (threads 128..191)
    if (tid < HD) {
      float a = sm.bf1[tid];
      for (int k = 0; k < ZD; ++k)  a += sm.z[k] * (float)sm.Wf1[k * HD + tid];
      for (int k = 0; k < RD; ++k)  a += sm.r[k] * (float)sm.Wf1[(ZD + k) * HD + tid];
      a += t * (float)sm.Wf1[192 * HD + tid];
      sm.f1a[tid] = siluf(a);
    } else if (tid < HD + ZD) {
      int k = tid - HD;
      float a = 0.f;
      for (int h = 0; h < HD; ++h) a += sm.S[h] * (float)sm.WTd1[h * 68 + k];
      float g = sm.z[k] + t * a;                       // prior grad z + t * dU_lik/dz
      sm.gv[k] = fminf(fmaxf(g, -100.f), 100.f);
    }
    __syncthreads();

    // P7b: drift f2 (threads 0..127) || re-zero S for next step (128..255)
    if (tid < HD) {
      float a = sm.bf2[tid];
      for (int k = 0; k < HD; ++k) a += sm.f1a[k] * (float)sm.Wf2[k * HD + tid];
      sm.f2a[tid] = siluf(a);
    } else {
      sm.S[tid - HD] = 0.f;
    }
    __syncthreads();

    // P7c: z update
    if (tid < ZD) {
      float bdr = sm.bf3[tid];
      for (int h = 0; h < HD; ++h) bdr += sm.f2a[h] * (float)sm.Wf3[h * ZD + tid];
      float nz = noises[((size_t)st * BTOT + (size_t)b) * ZD + tid];
      sm.z[tid] = sm.z[tid] + (bdr - sm.gv[tid]) * dt + dif * nz;
    }
    __syncthreads();
  }

  if (tid < ZD) out[(size_t)b * ZD + tid] = sm.z[tid];
}

extern "C" void kernel_launch(void* const* d_in, const int* in_sizes, int n_in,
                              void* d_out, int out_size, void* d_ws, size_t ws_size,
                              hipStream_t stream) {
  (void)in_sizes; (void)n_in; (void)d_ws; (void)ws_size; (void)out_size;
  const float* x_ctx = (const float*)d_in[0];
  const float* y_ctx = (const float*)d_in[1];
  const float* mask  = (const float*)d_in[2];
  const float* z0    = (const float*)d_in[3];
  const float* nois  = (const float*)d_in[4];
  const float* We1 = (const float*)d_in[5];  const float* be1 = (const float*)d_in[6];
  const float* We2 = (const float*)d_in[7];  const float* be2 = (const float*)d_in[8];
  const float* We3 = (const float*)d_in[9];  const float* be3 = (const float*)d_in[10];
  const float* Wd1 = (const float*)d_in[11]; const float* bd1 = (const float*)d_in[12];
  const float* Wd2 = (const float*)d_in[13]; const float* bd2 = (const float*)d_in[14];
  const float* Wd3 = (const float*)d_in[15]; const float* bd3 = (const float*)d_in[16];
  const float* Wf1 = (const float*)d_in[17]; const float* bf1 = (const float*)d_in[18];
  const float* Wf2 = (const float*)d_in[19]; const float* bf2 = (const float*)d_in[20];
  const float* Wf3 = (const float*)d_in[21]; const float* bf3 = (const float*)d_in[22];
  // d_in[23] = steps (int, ==20, baked in as NSTEPS)
  float* out = (float*)d_out;

  metanets_persistent_kernel<<<BTOT, 256, 0, stream>>>(
      x_ctx, y_ctx, mask, z0, nois,
      We1, be1, We2, be2, We3, be3,
      Wd1, bd1, Wd2, bd2, Wd3, bd3,
      Wf1, bf1, Wf2, bf2, Wf3, bf3,
      out);
}
